// MMD_loss_5360119185423
// MI455X (gfx1250) — matrix operand order repaired and checked by reading, not compiled, not run on hardware
//
#include <hip/hip_runtime.h>
#include <hip/hip_bf16.h>

typedef __attribute__((ext_vector_type(16))) _Float16 v16h;
typedef __attribute__((ext_vector_type(8)))  _Float16 v8h;
typedef __attribute__((ext_vector_type(8)))  float    v8f;

#define KDIM 128

// ---- Pass 1: f32 -> f16 convert + f32 row squared-norms ------------------
__global__ void mmd_prep_kernel(const float* __restrict__ X,
                                const float* __restrict__ Y,
                                int n1, int n2,
                                _Float16* __restrict__ Xh,
                                _Float16* __restrict__ Yh,
                                float* __restrict__ x2,
                                float* __restrict__ y2) {
    int r = blockIdx.x * blockDim.x + threadIdx.x;
    const float* src;
    _Float16* dst;
    float* nrm;
    int row;
    if (r < n1)            { src = X; dst = Xh; nrm = x2; row = r; }
    else if (r < n1 + n2)  { src = Y; dst = Yh; nrm = y2; row = r - n1; }
    else return;
    const float* p = src + (size_t)row * KDIM;
    _Float16*    q = dst + (size_t)row * KDIM;
    float s = 0.f;
#pragma unroll 8
    for (int k = 0; k < KDIM; ++k) {
        float v = p[k];
        s += v * v;
        q[k] = (_Float16)v;
    }
    nrm[row] = s;
}

// ---- WMMA operand loaders (per ISA 16-bit layouts, wave32) ---------------
// A 16x32: lane m, half h: elems 0..7 = K[k0+8h .. +7], elems 8..15 = K[k0+16+8h .. +7]
__device__ __forceinline__ v16h ld_a(const _Float16* __restrict__ row, int k0, int half) {
    const v8h* p = (const v8h*)row;
    v8h lo = p[(k0 >> 3) + half];
    v8h hi = p[(k0 >> 3) + 2 + half];
    v16h r;
#pragma unroll
    for (int i = 0; i < 8; ++i) { r[i] = lo[i]; r[i + 8] = hi[i]; }
    return r;
}
// B 32x16 (column data = row of the other operand): lane n, half h:
// elems 0..7 = K[k0+8h .. +7], elems 8..15 = K[k0+16+8h .. +7]
__device__ __forceinline__ v16h ld_b(const _Float16* __restrict__ row, int k0, int half) {
    const v8h* p = (const v8h*)row;
    v8h lo = p[(k0 >> 3) + half];
    v8h hi = p[(k0 >> 3) + 2 + half];
    v16h r;
#pragma unroll
    for (int i = 0; i < 8; ++i) { r[i] = lo[i]; r[i + 8] = hi[i]; }
    return r;
}

// ---- Pass 2: tiled Gram + exp + per-block partial sums -------------------
// Wave handles one 16-row strip of A against 4 adjacent 16-col tiles of B.
// grid.x = nB/64 tiles-of-4, grid.y = nA/128 (8 waves/block, 16 rows/wave)
__global__ __launch_bounds__(256)
void mmd_pair_kernel(const _Float16* __restrict__ A,
                     const _Float16* __restrict__ B,
                     const float* __restrict__ a2,
                     const float* __restrict__ b2,
                     float* __restrict__ partial,
                     float inv_mu) {
    const int lane  = threadIdx.x & 31;
    const int wave  = threadIdx.x >> 5;
    const int half  = lane >> 4;
    const int sub   = lane & 15;
    const int tileM  = blockIdx.y * 8 + wave;
    const int tileN0 = blockIdx.x * 4;

    const _Float16* arow  = A + (size_t)(tileM * 16 + sub) * KDIM;
    const _Float16* brow0 = B + (size_t)(tileN0 * 16 + sub) * KDIM;

    v8f c0 = {}, c1 = {}, c2 = {}, c3 = {};
#pragma unroll
    for (int k0 = 0; k0 < KDIM; k0 += 32) {
        v16h a = ld_a(arow, k0, half);
        v16h b;
        b  = ld_b(brow0 + (size_t)0 * 16 * KDIM, k0, half);
        c0 = __builtin_amdgcn_wmma_f32_16x16x32_f16(false, a, false, b, (short)0, c0, false, false);
        b  = ld_b(brow0 + (size_t)1 * 16 * KDIM, k0, half);
        c1 = __builtin_amdgcn_wmma_f32_16x16x32_f16(false, a, false, b, (short)0, c1, false, false);
        b  = ld_b(brow0 + (size_t)2 * 16 * KDIM, k0, half);
        c2 = __builtin_amdgcn_wmma_f32_16x16x32_f16(false, a, false, b, (short)0, c2, false, false);
        b  = ld_b(brow0 + (size_t)3 * 16 * KDIM, k0, half);
        c3 = __builtin_amdgcn_wmma_f32_16x16x32_f16(false, a, false, b, (short)0, c3, false, false);
    }

    // C layout: VGPR r -> M = r + 8*half, N = sub
    float x2v[8];
#pragma unroll
    for (int r = 0; r < 8; ++r) x2v[r] = a2[tileM * 16 + r + 8 * half];

    float s = 0.f;
#pragma unroll
    for (int j = 0; j < 4; ++j) {
        v8f c = (j == 0) ? c0 : (j == 1) ? c1 : (j == 2) ? c2 : c3;
        float y2n = b2[(tileN0 + j) * 16 + sub];
#pragma unroll
        for (int r = 0; r < 8; ++r) {
            float D = x2v[r] + y2n - 2.0f * c[r];
            s += __expf(-D * inv_mu);
        }
    }

    // wave32 reduction
#pragma unroll
    for (int o = 16; o > 0; o >>= 1) s += __shfl_xor(s, o, 32);

    __shared__ float red[8];
    if (lane == 0) red[wave] = s;
    __syncthreads();
    if (threadIdx.x == 0) {
        float t = 0.f;
#pragma unroll
        for (int i = 0; i < 8; ++i) t += red[i];
        partial[blockIdx.y * gridDim.x + blockIdx.x] = t;
    }
}

// ---- Pass 3: deterministic fixed-order final reduction -------------------
__global__ void mmd_finalize_kernel(const float* __restrict__ partial,
                                    int nxx, int nyy, int nxy,
                                    float a1, float a2c, float mbeta,
                                    float* __restrict__ out) {
    const float* pxx = partial;
    const float* pyy = partial + nxx;
    const float* pxy = partial + nxx + nyy;
    float s0 = 0.f, s1 = 0.f, s2 = 0.f;
    for (int i = threadIdx.x; i < nxx; i += blockDim.x) s0 += pxx[i];
    for (int i = threadIdx.x; i < nyy; i += blockDim.x) s1 += pyy[i];
    for (int i = threadIdx.x; i < nxy; i += blockDim.x) s2 += pxy[i];
    __shared__ float red[256];
    red[threadIdx.x] = a1 * s0 + a2c * s1 + mbeta * s2;
    __syncthreads();
    for (int o = 128; o > 0; o >>= 1) {
        if ((int)threadIdx.x < o) red[threadIdx.x] += red[threadIdx.x + o];
        __syncthreads();
    }
    if (threadIdx.x == 0) out[0] = red[0];
}

extern "C" void kernel_launch(void* const* d_in, const int* in_sizes, int n_in,
                              void* d_out, int out_size, void* d_ws, size_t ws_size,
                              hipStream_t stream) {
    (void)n_in; (void)out_size; (void)ws_size;
    const float* X = (const float*)d_in[0];
    const float* Y = (const float*)d_in[1];
    float* out = (float*)d_out;

    const int n1 = in_sizes[0] / KDIM;   // 4096
    const int n2 = in_sizes[1] / KDIM;   // 4096

    // workspace layout
    char* ws = (char*)d_ws;
    _Float16* Xh = (_Float16*)ws;            ws += (size_t)n1 * KDIM * sizeof(_Float16);
    _Float16* Yh = (_Float16*)ws;            ws += (size_t)n2 * KDIM * sizeof(_Float16);
    float* x2 = (float*)ws;                  ws += (size_t)n1 * sizeof(float);
    float* y2 = (float*)ws;                  ws += (size_t)n2 * sizeof(float);
    float* partial = (float*)ws;

    const float inv_mu = 0.5f;  // 1 / KERNEL_MU (2.0)
    const float a1    = (float)(1.0 / ((double)n1 * n1 - n1));
    const float a2c   = (float)(1.0 / ((double)n2 * n2 - n2));
    const float mbeta = (float)(-2.0 / ((double)n1 * (double)n2)); // kxy + kyx folded

    // pass 1: convert + norms
    {
        int total = n1 + n2;
        mmd_prep_kernel<<<(total + 255) / 256, 256, 0, stream>>>(X, Y, n1, n2, Xh, Yh, x2, y2);
    }

    // pass 2: three Gram passes
    dim3 blk(256);
    dim3 gXX(n1 / 64, n1 / 128);
    dim3 gYY(n2 / 64, n2 / 128);
    dim3 gXY(n2 / 64, n1 / 128);
    const int nxx = gXX.x * gXX.y;
    const int nyy = gYY.x * gYY.y;
    const int nxy = gXY.x * gXY.y;

    mmd_pair_kernel<<<gXX, blk, 0, stream>>>(Xh, Xh, x2, x2, partial,             inv_mu);
    mmd_pair_kernel<<<gYY, blk, 0, stream>>>(Yh, Yh, y2, y2, partial + nxx,       inv_mu);
    mmd_pair_kernel<<<gXY, blk, 0, stream>>>(Xh, Yh, x2, y2, partial + nxx + nyy, inv_mu);

    // pass 3: deterministic scalar reduction
    mmd_finalize_kernel<<<1, 256, 0, stream>>>(partial, nxx, nyy, nxy, a1, a2c, mbeta, out);
}
